// ProfilingHybridMoEWrapper_85993835200648
// MI455X (gfx1250) — matrix operand order repaired and checked by reading, not compiled, not run on hardware
//
#include <hip/hip_runtime.h>
#include <hip/hip_bf16.h>

typedef __attribute__((ext_vector_type(16))) __bf16 v16bf;
typedef __attribute__((ext_vector_type(8)))  __bf16 v8bf;
typedef __attribute__((ext_vector_type(8)))  float  v8f;
typedef __attribute__((ext_vector_type(4)))  int    v4i;

typedef __attribute__((address_space(1))) v4i* gv4i_p;   // global
typedef __attribute__((address_space(3))) v4i* lv4i_p;   // LDS

#define HID       2048
#define INTER     1408
#define TWO_INTER 2816
#define NE        8
#define NTOK      2048
#define NPAIR     (NTOK * 2)

#define MBLK 64
#define NBLK 128
#define KBLK 32
#define LDA  40   // padded LDS stride (halves): 80B rows -> 16B-aligned chunks
#define LDB  40

#if defined(__has_builtin)
#if __has_builtin(__builtin_amdgcn_global_load_async_to_lds_b128)
#define HAVE_ASYNC_LDS 1
#endif
#if __has_builtin(__builtin_amdgcn_s_wait_asynccnt)
#define HAVE_ASYNC_WAIT 1
#endif
#endif

__device__ __forceinline__ void async_copy16(const void* g, void* l) {
#ifdef HAVE_ASYNC_LDS
  __builtin_amdgcn_global_load_async_to_lds_b128(
      (gv4i_p)(void*)g, (lv4i_p)(void*)l, 0, 0);
#else
  *(uint4*)l = *(const uint4*)g;
#endif
}

template <int N>
__device__ __forceinline__ void async_wait() {
#ifdef HAVE_ASYNC_LDS
#ifdef HAVE_ASYNC_WAIT
  __builtin_amdgcn_s_wait_asynccnt(N);
#else
  asm volatile("s_wait_asynccnt %0" ::"i"(N) : "memory");
#endif
#endif
}

union FragAB { v16bf v; v8bf h[2]; };

// ---------------- fp32 -> bf16 bulk convert (hardware cvt) --------------
__global__ __launch_bounds__(256) void cvt_f32_bf16(
    const float* __restrict__ src, __bf16* __restrict__ dst, int n4) {
  int stride = gridDim.x * 256;
  for (int i = blockIdx.x * 256 + threadIdx.x; i < n4; i += stride) {
    float4 f = ((const float4*)src)[i];
    union { __bf16 b[4]; uint2 u; } o;
    o.b[0] = (__bf16)f.x; o.b[1] = (__bf16)f.y;
    o.b[2] = (__bf16)f.z; o.b[3] = (__bf16)f.w;
    ((uint2*)dst)[i] = o.u;
  }
}

// ---------------- Router: logits -> softmax -> top-2 --------------------
__global__ __launch_bounds__(256) void moe_router(
    const float* __restrict__ x, const float* __restrict__ gw,
    int* __restrict__ sel, float* __restrict__ selw) {
  __shared__ float red[NE][256];
  __shared__ float logits[NE];
  int t = blockIdx.x, tid = threadIdx.x;
  float acc[NE];
#pragma unroll
  for (int e = 0; e < NE; e++) acc[e] = 0.f;
  for (int h = tid; h < HID; h += 256) {
    float xv = x[(size_t)t * HID + h];
#pragma unroll
    for (int e = 0; e < NE; e++) acc[e] += xv * gw[e * HID + h];
  }
#pragma unroll
  for (int e = 0; e < NE; e++) red[e][tid] = acc[e];
  __syncthreads();
  if (tid < NE) {
    float s = 0.f;
    for (int i = 0; i < 256; i++) s += red[tid][i];
    logits[tid] = s;
  }
  __syncthreads();
  if (tid == 0) {
    float mx = logits[0];
    for (int e = 1; e < NE; e++) mx = fmaxf(mx, logits[e]);
    float p[NE], s = 0.f;
    for (int e = 0; e < NE; e++) { p[e] = __expf(logits[e] - mx); s += p[e]; }
    for (int e = 0; e < NE; e++) p[e] /= s;
    int i1 = 0;
    for (int e = 1; e < NE; e++) if (p[e] > p[i1]) i1 = e;
    int i2 = (i1 == 0) ? 1 : 0;
    for (int e = 0; e < NE; e++) if (e != i1 && p[e] > p[i2]) i2 = e;
    sel[t * 2 + 0] = i1; selw[t * 2 + 0] = p[i1];
    sel[t * 2 + 1] = i2; selw[t * 2 + 1] = p[i2];
  }
}

__global__ __launch_bounds__(256) void moe_count(
    const int* __restrict__ sel, int* __restrict__ offsets, int* __restrict__ cursor) {
  __shared__ int cnt[NE];
  int tid = threadIdx.x;
  if (tid < NE) cnt[tid] = 0;
  __syncthreads();
  for (int p = tid; p < NPAIR; p += 256) atomicAdd(&cnt[sel[p]], 1);
  __syncthreads();
  if (tid == 0) {
    int o = 0;
    for (int e = 0; e < NE; e++) { offsets[e] = o; cursor[e] = o; o += cnt[e]; }
    offsets[NE] = o;
  }
}

__global__ __launch_bounds__(256) void moe_scatter(
    const int* __restrict__ sel, const float* __restrict__ selw,
    int* __restrict__ cursor, int* __restrict__ tok_idx, float* __restrict__ pair_w) {
  int p = blockIdx.x * 256 + threadIdx.x;
  if (p >= NPAIR) return;
  int e = sel[p];
  int pos = atomicAdd(&cursor[e], 1);
  tok_idx[pos] = p >> 1;
  pair_w[pos]  = selw[p];
}

__global__ __launch_bounds__(256) void zero_out(float* __restrict__ out, int n) {
  for (int i = blockIdx.x * blockDim.x + threadIdx.x; i < n; i += gridDim.x * blockDim.x)
    out[i] = 0.f;
}

// ------- GEMM1 + fused SwiGLU: act = silu(X*Wg^T) * (X*Wu^T), bf16 WMMA --
__global__ __launch_bounds__(256) void moe_gemm1(
    const __bf16* __restrict__ xbf, const __bf16* __restrict__ wgu,
    const int* __restrict__ offsets, const int* __restrict__ tok_idx,
    __bf16* __restrict__ act) {
  const int MT = NTOK / MBLK;              // 32 M-tiles per expert (capacity)
  int e  = blockIdx.x / MT;
  int mt = blockIdx.x % MT;
  int base = offsets[e];
  int rows = offsets[e + 1] - base;
  int m0 = mt * MBLK;
  if (m0 >= rows) return;
  int n0g = blockIdx.y * 64;               // intermediate-col base (0..1344)

  __shared__ __bf16 sA[2][MBLK * LDA];
  __shared__ __bf16 sB[2][NBLK * LDB];
  __shared__ float  sC[MBLK * NBLK];       // f32 exchange for SwiGLU epilogue
  __shared__ int    sTok[MBLK];

  int tid = threadIdx.x;
  if (tid < MBLK) {
    int r = m0 + tid;
    sTok[tid] = tok_idx[base + ((r < rows) ? r : (rows - 1))];  // clamp
  }
  __syncthreads();

  // per-thread staging addresses
  int ar = tid >> 2, aq = tid & 3;                     // A: row, 8-half chunk
  const __bf16* gA = xbf + (size_t)sTok[ar] * HID + aq * 8;
  int aoff = ar * LDA + aq * 8;

  int br0 = tid >> 2, bq0 = tid & 3;                   // B chunk 0: rows 0..63
  int br1 = (tid + 256) >> 2, bq1 = tid & 3;           // B chunk 1: rows 64..127
  int bcol0 = n0g + br0;                               // gate half
  int bcol1 = INTER + n0g + (br1 - 64);                // up half
  const __bf16* wbase = wgu + (size_t)e * TWO_INTER * HID;
  const __bf16* gB0 = wbase + (size_t)bcol0 * HID + bq0 * 8;
  const __bf16* gB1 = wbase + (size_t)bcol1 * HID + bq1 * 8;
  int boff0 = br0 * LDB + bq0 * 8;
  int boff1 = br1 * LDB + bq1 * 8;

  int wave = tid >> 5, lane = tid & 31;
  int wm = wave >> 2, wn = wave & 3;                   // 2x4 waves, 32x32 each
  int lm = lane & 15, lg = lane >> 4;

  v8f acc[2][2];
#pragma unroll
  for (int a = 0; a < 2; a++)
#pragma unroll
    for (int b = 0; b < 2; b++)
#pragma unroll
      for (int g = 0; g < 8; g++) acc[a][b][g] = 0.f;

  const int NIT = HID / KBLK;                          // 64
  async_copy16(gA, &sA[0][aoff]);
  async_copy16(gB0, &sB[0][boff0]);
  async_copy16(gB1, &sB[0][boff1]);

  for (int it = 0; it < NIT; ++it) {
    int cur = it & 1;
    if (it + 1 < NIT) {
      int nxt = cur ^ 1, kb = (it + 1) * KBLK;
      async_copy16(gA + kb, &sA[nxt][aoff]);
      async_copy16(gB0 + kb, &sB[nxt][boff0]);
      async_copy16(gB1 + kb, &sB[nxt][boff1]);
      async_wait<3>();
    } else {
      async_wait<0>();
    }
    __syncthreads();

#pragma unroll
    for (int a = 0; a < 2; a++) {
      FragAB fa;
      int arr = (wm * 32 + a * 16 + lm) * LDA;
      fa.h[0] = *(const v8bf*)&sA[cur][arr + lg * 8];
      fa.h[1] = *(const v8bf*)&sA[cur][arr + 16 + lg * 8];
#pragma unroll
      for (int b = 0; b < 2; b++) {
        FragAB fb;
        int brr = (wn * 32 + b * 16 + lm) * LDB + lg * 8;
        fb.h[0] = *(const v8bf*)&sB[cur][brr];
        fb.h[1] = *(const v8bf*)&sB[cur][brr + 16];
        acc[a][b] = __builtin_amdgcn_wmma_f32_16x16x32_bf16(
            false, fa.v, false, fb.v, (short)0, acc[a][b], false, false);
      }
    }
    __syncthreads();
  }

  // epilogue: exchange via LDS, fuse SwiGLU, store bf16 act
#pragma unroll
  for (int a = 0; a < 2; a++)
#pragma unroll
    for (int b = 0; b < 2; b++)
#pragma unroll
      for (int g = 0; g < 8; g++) {
        int r = wm * 32 + a * 16 + lg * 8 + g;
        int c = wn * 32 + b * 16 + lm;
        sC[r * NBLK + c] = acc[a][b][g];
      }
  __syncthreads();
#pragma unroll
  for (int j = 0; j < 16; j++) {
    int elem = tid * 16 + j;                 // 64x64 outputs
    int r = elem >> 6, c = elem & 63;
    if (m0 + r < rows) {
      float gv = sC[r * NBLK + c];
      float uv = sC[r * NBLK + 64 + c];
      float sv = gv / (1.f + __expf(-gv));
      act[(size_t)(base + m0 + r) * INTER + n0g + c] = (__bf16)(sv * uv);
    }
  }
}

// ------- GEMM2: out[t] += w * (act * Wdown^T), bf16 WMMA, atomic combine -
__global__ __launch_bounds__(256) void moe_gemm2(
    const __bf16* __restrict__ act, const __bf16* __restrict__ wdn,
    const int* __restrict__ offsets, const int* __restrict__ tok_idx,
    const float* __restrict__ pair_w, float* __restrict__ out) {
  const int MT = NTOK / MBLK;
  int e  = blockIdx.x / MT;
  int mt = blockIdx.x % MT;
  int base = offsets[e];
  int rows = offsets[e + 1] - base;
  int m0 = mt * MBLK;
  if (m0 >= rows) return;
  int n0 = blockIdx.y * NBLK;              // over HID, 16 tiles

  __shared__ __bf16 sA[2][MBLK * LDA];
  __shared__ __bf16 sB[2][NBLK * LDB];
  __shared__ int    sTok[MBLK];
  __shared__ float  sW[MBLK];

  int tid = threadIdx.x;
  if (tid < MBLK) {
    int r = m0 + tid;
    sTok[tid] = tok_idx[base + ((r < rows) ? r : (rows - 1))];
    sW[tid]   = (r < rows) ? pair_w[base + r] : 0.f;
  }
  __syncthreads();

  int ar = tid >> 2, aq = tid & 3;
  int arow = m0 + ar; if (arow >= rows) arow = rows - 1;          // clamp
  const __bf16* gA = act + (size_t)(base + arow) * INTER + aq * 8;
  int aoff = ar * LDA + aq * 8;

  int br0 = tid >> 2, br1 = (tid + 256) >> 2, bq = tid & 3;
  const __bf16* wbase = wdn + (size_t)e * HID * INTER;
  const __bf16* gB0 = wbase + (size_t)(n0 + br0) * INTER + bq * 8;
  const __bf16* gB1 = wbase + (size_t)(n0 + br1) * INTER + bq * 8;
  int boff0 = br0 * LDB + bq * 8;
  int boff1 = br1 * LDB + bq * 8;

  int wave = tid >> 5, lane = tid & 31;
  int wm = wave >> 2, wn = wave & 3;
  int lm = lane & 15, lg = lane >> 4;

  v8f acc[2][2];
#pragma unroll
  for (int a = 0; a < 2; a++)
#pragma unroll
    for (int b = 0; b < 2; b++)
#pragma unroll
      for (int g = 0; g < 8; g++) acc[a][b][g] = 0.f;

  const int NIT = INTER / KBLK;            // 44
  async_copy16(gA, &sA[0][aoff]);
  async_copy16(gB0, &sB[0][boff0]);
  async_copy16(gB1, &sB[0][boff1]);

  for (int it = 0; it < NIT; ++it) {
    int cur = it & 1;
    if (it + 1 < NIT) {
      int nxt = cur ^ 1, kb = (it + 1) * KBLK;
      async_copy16(gA + kb, &sA[nxt][aoff]);
      async_copy16(gB0 + kb, &sB[nxt][boff0]);
      async_copy16(gB1 + kb, &sB[nxt][boff1]);
      async_wait<3>();
    } else {
      async_wait<0>();
    }
    __syncthreads();

#pragma unroll
    for (int a = 0; a < 2; a++) {
      FragAB fa;
      int arr = (wm * 32 + a * 16 + lm) * LDA;
      fa.h[0] = *(const v8bf*)&sA[cur][arr + lg * 8];
      fa.h[1] = *(const v8bf*)&sA[cur][arr + 16 + lg * 8];
#pragma unroll
      for (int b = 0; b < 2; b++) {
        FragAB fb;
        int brr = (wn * 32 + b * 16 + lm) * LDB + lg * 8;
        fb.h[0] = *(const v8bf*)&sB[cur][brr];
        fb.h[1] = *(const v8bf*)&sB[cur][brr + 16];
        acc[a][b] = __builtin_amdgcn_wmma_f32_16x16x32_bf16(
            false, fa.v, false, fb.v, (short)0, acc[a][b], false, false);
      }
    }
    __syncthreads();
  }

#pragma unroll
  for (int a = 0; a < 2; a++)
#pragma unroll
    for (int b = 0; b < 2; b++)
#pragma unroll
      for (int g = 0; g < 8; g++) {
        int lrow = wm * 32 + a * 16 + lg * 8 + g;
        if (m0 + lrow < rows) {
          int t = sTok[lrow];
          float w = sW[lrow];
          int col = n0 + wn * 32 + b * 16 + lm;
          atomicAdd(&out[(size_t)t * HID + col], w * acc[a][b][g]);
        }
      }
}

// ---------------- Host-side launch --------------------------------------
extern "C" void kernel_launch(void* const* d_in, const int* in_sizes, int n_in,
                              void* d_out, int out_size, void* d_ws, size_t ws_size,
                              hipStream_t stream) {
  const float* x   = (const float*)d_in[0];   // [1, 2048, 2048]
  const float* gw  = (const float*)d_in[1];   // [8, 2048]
  const float* wgu = (const float*)d_in[2];   // [8, 2816, 2048]
  const float* wdn = (const float*)d_in[3];   // [8, 2048, 1408]
  float* out = (float*)d_out;                 // [1, 2048, 2048]

  char* ws = (char*)d_ws;
  size_t off = 0;
  auto carve = [&](size_t bytes) -> char* {
    char* p = ws + off;
    off += (bytes + 255) & ~(size_t)255;
    return p;
  };
  int*   sel     = (int*)  carve((size_t)NPAIR * sizeof(int));
  float* selw    = (float*)carve((size_t)NPAIR * sizeof(float));
  int*   offsets = (int*)  carve((NE + 1) * sizeof(int));
  int*   cursor  = (int*)  carve(NE * sizeof(int));
  int*   tok_idx = (int*)  carve((size_t)NPAIR * sizeof(int));
  float* pair_w  = (float*)carve((size_t)NPAIR * sizeof(float));
  __bf16* xbf  = (__bf16*)carve((size_t)NTOK * HID * 2);
  __bf16* wgub = (__bf16*)carve((size_t)NE * TWO_INTER * HID * 2);
  __bf16* wdnb = (__bf16*)carve((size_t)NE * HID * INTER * 2);
  __bf16* actb = (__bf16*)carve((size_t)NPAIR * INTER * 2);

  moe_router<<<NTOK, 256, 0, stream>>>(x, gw, sel, selw);
  moe_count<<<1, 256, 0, stream>>>(sel, offsets, cursor);
  moe_scatter<<<(NPAIR + 255) / 256, 256, 0, stream>>>(sel, selw, cursor, tok_idx, pair_w);
  zero_out<<<1024, 256, 0, stream>>>(out, NTOK * HID);

  cvt_f32_bf16<<<2048, 256, 0, stream>>>(x, xbf, NTOK * HID / 4);
  cvt_f32_bf16<<<2048, 256, 0, stream>>>(wgu, wgub, NE * TWO_INTER * HID / 4);
  cvt_f32_bf16<<<2048, 256, 0, stream>>>(wdn, wdnb, NE * HID * INTER / 4);

  dim3 g1(NE * (NTOK / MBLK), INTER / 64);         // (256, 22)
  moe_gemm1<<<g1, 256, 0, stream>>>(xbf, wgub, offsets, tok_idx, actb);

  dim3 g2(NE * (NTOK / MBLK), HID / NBLK);         // (256, 16)
  moe_gemm2<<<g2, 256, 0, stream>>>(actb, wdnb, offsets, tok_idx, pair_w, out);
}
